// RNN_MultiRegional_7155415515601
// MI455X (gfx1250) — matrix operand order repaired and checked by reading, not compiled, not run on hardware
//
#include <hip/hip_runtime.h>
#include <hip/hip_bf16.h>

typedef __bf16 bf16_t;
typedef __attribute__((ext_vector_type(16))) __bf16 bf16x16;
typedef __attribute__((ext_vector_type(8)))  __bf16 bf16x8;
typedef __attribute__((ext_vector_type(8)))  float  f32x8;
typedef int v4i __attribute__((vector_size(16)));   // matches builtin param type

#define H       256
#define H5      1280
#define BATCH   64
#define TSTEPS  512
#define NWG     40
#define LDSP    264   /* 256 + 8 bf16 pad -> fragment ds_loads spread across banks */
#define NEXC    180   /* H - int(0.3*H) = 256 - 76 */

#define AS1 __attribute__((address_space(1)))
#define AS3 __attribute__((address_space(3)))

#if __has_builtin(__builtin_amdgcn_global_load_async_to_lds_b128)
#define HAVE_ASYNC_LDS 1
#else
#define HAVE_ASYNC_LDS 0
#endif

__device__ __forceinline__ void wait_async_zero() {
#if __has_builtin(__builtin_amdgcn_s_wait_asynccnt)
  __builtin_amdgcn_s_wait_asynccnt(0);
#else
  asm volatile("s_wait_asynccnt 0x0" ::: "memory");
#endif
}

__device__ __forceinline__ float ht(float w) {          // hardtanh(w, 1e-10, 1)
  return fminf(fmaxf(w, 1e-10f), 1.0f);
}

// Per output region r, the 3 source column-blocks (padded with blocks that are
// identically zero in W so every wave runs the same uniform 24-WMMA body).
static __device__ const int SRC3[5][3] = {
  {0, 3, 4},   // str  <- str2str(-fixed), thal2str, m12str
  {0, 1, 2},   // stn  <- str2stn   (1,2 zero pad)
  {0, 1, 2},   // snr  <- str2snr, stn2snr(=str2stn quirk)  (2 zero pad)
  {2, 4, 0},   // thal <- snr2thal, m12thal  (0 zero pad)
  {3, 4, 0},   // m1   <- thal2m1, m12m1     (0 zero pad)
};

// ---------------------------------------------------------------------------
// Kernel 1: materialize W_rec (row-major [n][k], n=output neuron) as bf16.
// ---------------------------------------------------------------------------
__global__ void build_w_kernel(const float* __restrict__ w_str2str,
                               const float* __restrict__ w_str2snr,
                               const float* __restrict__ w_m12m1,
                               const float* __restrict__ w_m12str,
                               const float* __restrict__ w_thal2m1,
                               const float* __restrict__ w_m12thal,
                               const float* __restrict__ w_thal2str,
                               const float* __restrict__ w_str2stn,
                               const float* __restrict__ w_snr2thal,
                               const float* __restrict__ fixedw,
                               bf16_t* __restrict__ Wb) {
  int idx = blockIdx.x * blockDim.x + threadIdx.x;
  if (idx >= H5 * H5) return;
  int n = idx / H5, k = idx - n * H5;
  int r = n >> 8, c = k >> 8, i = n & 255, j = k & 255;
  int o = i * H + j;
  float v = 0.0f;
  if (r == 0) {
    if (c == 0)      v = -fixedw[o];                       // str2str_mask==0
    else if (c == 3) v = ht(w_thal2str[o]);
    else if (c == 4) v = (j < NEXC) ? ht(w_m12str[o]) : 0.0f;
  } else if (r == 1) {
    if (c == 0)      v = (j >= H / 2) ? ht(w_str2stn[o]) : 0.0f;
  } else if (r == 2) {
    if (c == 0)      v = (j < H / 2) ? -ht(w_str2snr[o]) : 0.0f;
    else if (c == 1) v = -ht(w_str2stn[o]);                // faithful quirk
  } else if (r == 3) {
    if (c == 2)      v = -ht(w_snr2thal[o]);
    else if (c == 4) v = ht(w_m12thal[o]);
  } else {
    if (c == 3)      v = ht(w_thal2m1[o]);
    else if (c == 4) v = ht(w_m12m1[o]) * ((j < NEXC) ? 1.0f : -1.0f);
  }
  Wb[idx] = (bf16_t)v;
}

// ---------------------------------------------------------------------------
// Kernel 2: init state buffers, barrier words, and constant mean/std outputs.
// ---------------------------------------------------------------------------
__global__ void init_kernel(const float* __restrict__ hn,
                            const float* __restrict__ mean_b,
                            const float* __restrict__ std_b,
                            float* __restrict__ hf, bf16_t* __restrict__ hb,
                            float* __restrict__ out, unsigned* __restrict__ bar) {
  int idx = blockIdx.x * blockDim.x + threadIdx.x;
  if (idx < BATCH * H5) {
    float v = hn[idx];
    hf[idx] = v;                 hb[idx] = (bf16_t)v;
    hf[BATCH * H5 + idx] = 0.f;  hb[BATCH * H5 + idx] = (bf16_t)0.0f;
  }
  if (idx < BATCH * TSTEPS) {    // m1_mask == 0 -> outputs are just the biases
    out[idx] = mean_b[0];
    out[BATCH * TSTEPS + idx] = fminf(fmaxf(std_b[0], -5.0f), 10.0f);
  }
  if (idx == 0) { bar[0] = 0u; bar[1] = 0u; }
}

// ---------------------------------------------------------------------------
// Kernel 3: persistent recurrence. 40 WGs x 256 threads; wave = one 16x16 tile.
// ---------------------------------------------------------------------------
__global__ __launch_bounds__(256, 1)
void rnn_step_kernel(const bf16_t* __restrict__ Wb,
                     const float*  __restrict__ inp,    // [B][T][4]
                     const float*  __restrict__ inpw,   // [4][1280]
                     float*        __restrict__ hf,     // [2][B*H5] f32 master state
                     bf16_t*       __restrict__ hb,     // [2][B*H5] bf16 mirror
                     float*        __restrict__ out,
                     unsigned*     __restrict__ bar) {
  __shared__ bf16_t Alds[64 * LDSP];

  const int tid   = threadIdx.x;
  const int wg    = blockIdx.x;
  const int lane  = tid & 31;
  const int wave  = tid >> 5;
  const int m0    = (wave & 3) * 16;            // batch-row tile
  const int ntile = wg * 2 + (wave >> 2);       // 80 column tiles
  const int n0    = ntile * 16;
  const int region   = ntile >> 4;              // 16 n-tiles per region
  const int regionWG = wg >> 3;                 // same region for whole WG

  union BF { bf16x16 v; bf16x8 h[2]; };

  // ---- B fragments: W_rec^T tiles, register-resident for all 512 steps ----
  // B layout (16-bit 32x16): lane<16 holds col n0+lane, K[k0,k0+16);
  //                          lane>=16 holds col n0+lane-16, K[k0+16,k0+32).
  bf16x16 Bf[24];
  {
    const int ncol  = n0 + (lane & 15);
    const int khalf = (lane >> 4) * 16;
#pragma unroll
    for (int s = 0; s < 3; ++s) {
      const int cb = SRC3[region][s];
#pragma unroll
      for (int kf = 0; kf < 8; ++kf) {
        const bf16_t* p = Wb + (size_t)ncol * H5 + cb * H + kf * 32 + khalf;
        BF t; t.h[0] = *(const bf16x8*)p; t.h[1] = *(const bf16x8*)(p + 8);
        Bf[s * 8 + kf] = t.v;
      }
    }
  }

  // per-lane hardtanh'd input weights (thal drive), constant over t
  const int  ncol = n0 + (lane & 15);
  const float uw0 = ht(inpw[0 * H5 + ncol]);
  const float uw1 = ht(inpw[1 * H5 + ncol]);
  const float uw2 = ht(inpw[2 * H5 + ncol]);
  const float uw3 = ht(inpw[3 * H5 + ncol]);
  const float useU = (region == 3) ? 1.0f : 0.0f;
  const float4* inp4 = (const float4*)inp;

  float* rnn   = out + 2 * BATCH * TSTEPS;
  float* hlast = rnn + (size_t)BATCH * TSTEPS * H5;

  for (int t = 0; t < TSTEPS; ++t) {
    const int p = t & 1, q = p ^ 1;
    const bf16_t* hread_b  = hb + (size_t)p * (BATCH * H5);
    const float*  hread_f  = hf + (size_t)p * (BATCH * H5);
    float*        hwrite_f = hf + (size_t)q * (BATCH * H5);
    bf16_t*       hwrite_b = hb + (size_t)q * (BATCH * H5);

    f32x8 acc = {0.f, 0.f, 0.f, 0.f, 0.f, 0.f, 0.f, 0.f};

#pragma unroll
    for (int s = 0; s < 3; ++s) {
      const int cb = SRC3[regionWG][s];
      __syncthreads();                      // protect LDS reuse from prev block
      // cooperative stage: 64 rows x 256 bf16 of source block cb -> LDS
#pragma unroll
      for (int it = 0; it < 8; ++it) {
        int idx = it * 256 + tid;           // 2048 chunks of 8 bf16
        int row = idx >> 5, ch = idx & 31;
        const bf16_t* src = hread_b + row * H5 + cb * H + ch * 8;
        bf16_t*       dst = &Alds[row * LDSP + ch * 8];
#if HAVE_ASYNC_LDS
        // gfx1250 async copy: LDS written directly, tracked by ASYNCcnt
        __builtin_amdgcn_global_load_async_to_lds_b128(
            (AS1 v4i*)(uintptr_t)src,
            (AS3 v4i*)(uint32_t)(uintptr_t)dst, 0, 0);
#else
        *(bf16x8*)dst = *(const bf16x8*)src;
#endif
      }
#if HAVE_ASYNC_LDS
      wait_async_zero();   // drain own copies BEFORE barrier -> WG sees all
#endif
      __syncthreads();
      // A layout (16-bit 16x32): lane holds row m0+(lane&15),
      // K range [kb,kb+16) with kb = kf*32 + (lane>>4)*16 (matches B).
      const int arow = m0 + (lane & 15);
      const bf16_t* abase = &Alds[arow * LDSP + (lane >> 4) * 16];
#pragma unroll
      for (int kf = 0; kf < 8; ++kf) {
        BF a;
        a.h[0] = *(const bf16x8*)(abase + kf * 32);
        a.h[1] = *(const bf16x8*)(abase + kf * 32 + 8);
        acc = __builtin_amdgcn_wmma_f32_16x16x32_bf16(
            false, a.v, false, Bf[s * 8 + kf], (short)0, acc, false, false);
      }
    }

    // epilogue: h_new = relu(0.9 h + 0.1 (acc + u)); f32 master, bf16 mirror
#pragma unroll
    for (int i = 0; i < 8; ++i) {
      const int m = m0 + (lane >> 4) * 8 + i;       // C/D layout, VGPR i
      float4 x = inp4[m * TSTEPS + t];
      float u  = useU * (x.x * uw0 + x.y * uw1 + x.z * uw2 + x.w * uw3);
      float ho = hread_f[m * H5 + ncol];
      float hn = fmaxf(0.9f * ho + 0.1f * (acc[i] + u), 0.0f);
      hwrite_f[m * H5 + ncol] = hn;
      hwrite_b[m * H5 + ncol] = (bf16_t)hn;
      // streaming 168MB output: non-temporal so it doesn't evict L2-hot W/h
      __builtin_nontemporal_store(hn, &rnn[(size_t)m * (TSTEPS * H5) +
                                           (size_t)t * H5 + ncol]);
      if (t == TSTEPS - 1) hlast[m * H5 + ncol] = hn;
    }

    // device-wide sense-reversing barrier (40 co-resident WGs)
    __syncthreads();
    if (tid == 0) {
      __threadfence();
      unsigned e = __hip_atomic_load(&bar[1], __ATOMIC_ACQUIRE, __HIP_MEMORY_SCOPE_AGENT);
      unsigned a = __hip_atomic_fetch_add(&bar[0], 1u, __ATOMIC_ACQ_REL, __HIP_MEMORY_SCOPE_AGENT);
      if (a == NWG - 1) {
        __hip_atomic_store(&bar[0], 0u, __ATOMIC_RELAXED, __HIP_MEMORY_SCOPE_AGENT);
        __hip_atomic_fetch_add(&bar[1], 1u, __ATOMIC_ACQ_REL, __HIP_MEMORY_SCOPE_AGENT);
      } else {
        while (__hip_atomic_load(&bar[1], __ATOMIC_ACQUIRE, __HIP_MEMORY_SCOPE_AGENT) == e)
          __builtin_amdgcn_s_sleep(2);
      }
      __threadfence();
    }
    __syncthreads();
  }
}

// ---------------------------------------------------------------------------
extern "C" void kernel_launch(void* const* d_in, const int* in_sizes, int n_in,
                              void* d_out, int out_size, void* d_ws, size_t ws_size,
                              hipStream_t stream) {
  (void)in_sizes; (void)n_in; (void)out_size; (void)ws_size;
  const float* inp        = (const float*)d_in[0];
  const float* hn         = (const float*)d_in[1];
  const float* w_str2str  = (const float*)d_in[2];
  const float* w_str2snr  = (const float*)d_in[3];
  const float* w_m12m1    = (const float*)d_in[4];
  const float* w_m12str   = (const float*)d_in[5];
  const float* w_thal2m1  = (const float*)d_in[6];
  const float* w_m12thal  = (const float*)d_in[7];
  const float* w_thal2str = (const float*)d_in[8];
  const float* w_str2stn  = (const float*)d_in[9];
  /* d_in[10] w_stn2snr unused: reference uses w_str2stn for stn2snr */
  const float* w_snr2thal = (const float*)d_in[11];
  const float* fixedw     = (const float*)d_in[12];
  const float* inpw       = (const float*)d_in[13];
  const float* mean_b     = (const float*)d_in[15];
  const float* std_b      = (const float*)d_in[17];
  float* out = (float*)d_out;

  char* ws = (char*)d_ws;
  bf16_t*   Wb  = (bf16_t*)ws;                                  // 3,276,800 B
  float*    hf  = (float*)(ws + 3276800);                       //   655,360 B
  bf16_t*   hb  = (bf16_t*)(ws + 3276800 + 655360);             //   327,680 B
  unsigned* bar = (unsigned*)(ws + 3276800 + 655360 + 327680);  //         8 B

  build_w_kernel<<<(H5 * H5 + 255) / 256, 256, 0, stream>>>(
      w_str2str, w_str2snr, w_m12m1, w_m12str, w_thal2m1, w_m12thal,
      w_thal2str, w_str2stn, w_snr2thal, fixedw, Wb);
  init_kernel<<<(BATCH * H5 + 255) / 256, 256, 0, stream>>>(
      hn, mean_b, std_b, hf, hb, out, bar);
  rnn_step_kernel<<<NWG, 256, 0, stream>>>(Wb, inp, inpw, hf, hb, out, bar);
}
